// FrameAugment_43482248905048
// MI455X (gfx1250) — matrix operand order repaired and checked by reading, not compiled, not run on hardware
//
#include <hip/hip_runtime.h>
#include <hip/hip_bf16.h>

// Problem constants (match reference)
#define BB 32
#define NN 1024
#define FF 128
#define INV_T 5.0f          // 1 / 0.2
#define EPSV 1e-12f
#define ITERS 10

typedef __attribute__((ext_vector_type(16))) __bf16 v16bf;
typedef __attribute__((ext_vector_type(8)))  __bf16 v8bf;
typedef __attribute__((ext_vector_type(8)))  float  v8f;

// ---------------------------------------------------------------------------
// 1) m = (template + gumbel(u)) / T   (vectorized float4, template broadcast)
// ---------------------------------------------------------------------------
__global__ __launch_bounds__(256)
void gs_gumbel_init(const float* __restrict__ tmpl,
                    const float* __restrict__ u,
                    float* __restrict__ m) {
    const size_t TOT4 = (size_t)BB * NN * NN / 4;
    size_t i = (size_t)blockIdx.x * 256 + threadIdx.x;
    if (i >= TOT4) return;
    const float4 uu = ((const float4*)u)[i];
    const size_t rem4 = i % ((size_t)NN * NN / 4);
    const float4 tt = ((const float4*)tmpl)[rem4];
    float4 o;
    o.x = (tt.x + (-__logf(-__logf(uu.x + EPSV) + EPSV))) * INV_T;
    o.y = (tt.y + (-__logf(-__logf(uu.y + EPSV) + EPSV))) * INV_T;
    o.z = (tt.z + (-__logf(-__logf(uu.z + EPSV) + EPSV))) * INV_T;
    o.w = (tt.w + (-__logf(-__logf(uu.w + EPSV) + EPSV))) * INV_T;
    ((float4*)m)[i] = o;
}

// ---------------------------------------------------------------------------
// 2) feature f32 -> bf16 (done once; B operand of the WMMA GEMM)
// ---------------------------------------------------------------------------
__global__ __launch_bounds__(256)
void gs_f2bf(const float* __restrict__ f, __bf16* __restrict__ o) {
    size_t i = (size_t)blockIdx.x * 256 + threadIdx.x;
    if (i < (size_t)BB * NN * FF) o[i] = (__bf16)f[i];
}

// ---------------------------------------------------------------------------
// 3) zero per-column exp-sums
// ---------------------------------------------------------------------------
__global__ __launch_bounds__(256)
void gs_zero(float* __restrict__ p, int n) {
    int i = blockIdx.x * 256 + threadIdx.x;
    if (i < n) p[i] = 0.0f;
}

// ---------------------------------------------------------------------------
// 3b) invert column sums in place: csum <- 1/csum  (32K elems, ~free)
//     Lets all consumers use a single v_mul_f32 instead of an IEEE div chain.
// ---------------------------------------------------------------------------
__global__ __launch_bounds__(256)
void gs_rcp(float* __restrict__ p, int n) {
    int i = blockIdx.x * 256 + threadIdx.x;
    if (i < n) p[i] = 1.0f / p[i];
}

// ---------------------------------------------------------------------------
// 4) row softmax; optionally applies previous column softmax on load:
//    x <- exp(x) * inv_csum[j]  (no max needed for col stage: x in (0,1])
//    One block per row; 4 elements per thread held in registers.
// ---------------------------------------------------------------------------
__global__ __launch_bounds__(256)
void gs_row_softmax(const float* __restrict__ src, float* __restrict__ dst,
                    const float* __restrict__ icsum, int apply_col) {
    const int row = blockIdx.x;          // 0 .. B*N-1
    const int b = row >> 10;             // row / N
    const int tid = threadIdx.x;
    const float* rp = src + (size_t)row * NN;

    float x[4];
#pragma unroll
    for (int k = 0; k < 4; ++k) x[k] = rp[tid + k * 256];
    if (apply_col) {
        const float* cs = icsum + (size_t)b * NN;
#pragma unroll
        for (int k = 0; k < 4; ++k)
            x[k] = __expf(x[k]) * cs[tid + k * 256];
    }

    __shared__ float red_mx[8];
    __shared__ float red_sm[8];

    // --- max reduction (wave32 shuffles + cross-wave LDS) ---
    float mx = fmaxf(fmaxf(x[0], x[1]), fmaxf(x[2], x[3]));
#pragma unroll
    for (int off = 16; off > 0; off >>= 1)
        mx = fmaxf(mx, __shfl_xor(mx, off, 32));
    if ((tid & 31) == 0) red_mx[tid >> 5] = mx;
    __syncthreads();
    float gmx = red_mx[0];
#pragma unroll
    for (int w = 1; w < 8; ++w) gmx = fmaxf(gmx, red_mx[w]);

    // --- exp + sum reduction ---
    float s = 0.0f;
#pragma unroll
    for (int k = 0; k < 4; ++k) { x[k] = __expf(x[k] - gmx); s += x[k]; }
#pragma unroll
    for (int off = 16; off > 0; off >>= 1)
        s += __shfl_xor(s, off, 32);
    if ((tid & 31) == 0) red_sm[tid >> 5] = s;
    __syncthreads();
    float tot = 0.0f;
#pragma unroll
    for (int w = 0; w < 8; ++w) tot += red_sm[w];
    const float inv = 1.0f / tot;

    float* wp = dst + (size_t)row * NN;
#pragma unroll
    for (int k = 0; k < 4; ++k) wp[tid + k * 256] = x[k] * inv;
}

// ---------------------------------------------------------------------------
// 5) column stats: csum[b][j] += sum over a 256-row chunk of exp(m[b][r][j])
//    (inputs are row-softmax outputs in (0,1] -> max-free is numerically safe)
// ---------------------------------------------------------------------------
__global__ __launch_bounds__(256)
void gs_col_stats(const float* __restrict__ m, float* __restrict__ csum) {
    const int j  = blockIdx.x * 256 + threadIdx.x;
    const int r0 = blockIdx.y * 256;
    const int b  = blockIdx.z;
    const float* p = m + ((size_t)b * NN + r0) * NN + j;
    float s = 0.0f;
#pragma unroll 4
    for (int r = 0; r < 256; ++r) s += __expf(p[(size_t)r * NN]);
    atomicAdd(&csum[(size_t)b * NN + j], s);
}

// ---------------------------------------------------------------------------
// 6) out[b] = colsoftmax(m[b]) @ feature[b]  via v_wmma_f32_16x16x32_bf16
//    Block = 256 threads = 8 waves; covers 16 output rows x 128 cols.
//    Final column softmax (exp(x)*inv_csum[j]) fused into the A-tile load.
// ---------------------------------------------------------------------------
__global__ __launch_bounds__(256)
void gs_gemm_wmma(const float* __restrict__ m, const float* __restrict__ icsum,
                  const __bf16* __restrict__ fbf, float* __restrict__ out) {
    const int b  = blockIdx.y;
    const int i0 = blockIdx.x * 16;          // output-row tile
    const int tid  = threadIdx.x;
    const int lane = tid & 31;
    const int wv   = tid >> 5;               // 8 waves
    const int n0   = wv * 16;                // each wave: 16 output cols

    __shared__ __align__(16) __bf16 sA[16][32];       // [i_local][k_local]
    __shared__ __align__(16) __bf16 sB[128][40];      // transposed [n][k], padded row

    const float*  mrow = m    + ((size_t)b * NN + i0) * NN;
    const float*  cs   = icsum + (size_t)b * NN;
    const __bf16* fb   = fbf  + (size_t)b * NN * FF;

    v8f acc = {};

    for (int kk = 0; kk < NN; kk += 32) {
        // stage A (16x32) with fused column softmax + bf16 convert
#pragma unroll
        for (int t = tid; t < 512; t += 256) {
            const int r = t >> 5, c = t & 31;
            const float x = mrow[(size_t)r * NN + kk + c];
            sA[r][c] = (__bf16)(__expf(x) * cs[kk + c]);
        }
        // stage B (32x128) transposed into [n][k]; 32-bit global loads
#pragma unroll
        for (int t = tid; t < 2048; t += 256) {
            const int k = t >> 6, n2 = (t & 63) * 2;
            const __hip_bfloat162 v =
                *(const __hip_bfloat162*)&fb[(size_t)(kk + k) * FF + n2];
            sB[n2][k]     = *(const __bf16*)&v.x;
            sB[n2 + 1][k] = *(const __bf16*)&v.y;
        }
        __syncthreads();

        // A fragment (16x32 bf16, ISA layout):
        // lane<16 : row=lane,    K = {0..7, 16..23}
        // lane>=16: row=lane-16, K = {8..15, 24..31}
        const int ar = lane & 15;
        const int ao = (lane >> 4) * 8;
        v8bf alo = *(const v8bf*)&sA[ar][ao];
        v8bf ahi = *(const v8bf*)&sA[ar][ao + 16];
        v16bf a = __builtin_shufflevector(alo, ahi,
                    0,1,2,3,4,5,6,7,8,9,10,11,12,13,14,15);

        // B fragment (32x16 bf16): lane%16 = N, K = {0..7,16..23} / {8..15,24..31}
        const int bn = n0 + (lane & 15);
        const int bk = (lane >> 4) * 8;
        v8bf blo = *(const v8bf*)&sB[bn][bk];
        v8bf bhi = *(const v8bf*)&sB[bn][bk + 16];
        v16bf bf = __builtin_shufflevector(blo, bhi,
                    0,1,2,3,4,5,6,7,8,9,10,11,12,13,14,15);

        acc = __builtin_amdgcn_wmma_f32_16x16x32_bf16(
                  false, a, false, bf, (short)0, acc, false, false);
        __syncthreads();
    }

    // C layout: VGPR r -> (M=r, N=lane) for lanes 0-15, (M=r+8, N=lane-16) else
    const int cm = (lane >> 4) * 8;
    const int cn = n0 + (lane & 15);
    float* op = out + ((size_t)b * NN + i0) * FF;
#pragma unroll
    for (int r = 0; r < 8; ++r)
        op[(size_t)(r + cm) * FF + cn] = acc[r];
}

// ---------------------------------------------------------------------------
extern "C" void kernel_launch(void* const* d_in, const int* in_sizes, int n_in,
                              void* d_out, int out_size, void* d_ws, size_t ws_size,
                              hipStream_t stream) {
    (void)in_sizes; (void)n_in; (void)out_size; (void)ws_size;
    const float* feature = (const float*)d_in[0];   // [B,N,F]
    const float* tmpl    = (const float*)d_in[1];   // [1,N,N]
    const float* unif    = (const float*)d_in[2];   // [B,N,N]
    float* out = (float*)d_out;                     // [B,N,F]

    char* ws = (char*)d_ws;
    const size_t M_BYTES  = (size_t)BB * NN * NN * sizeof(float);   // 128 MB
    const size_t CS_BYTES = (size_t)BB * NN * sizeof(float);        // 128 KB
    float*  m    = (float*)ws;
    float*  csum = (float*)(ws + M_BYTES);
    __bf16* fbf  = (__bf16*)(ws + M_BYTES + CS_BYTES);

    const int CS_N = BB * NN;

    // init m and bf16 feature
    gs_gumbel_init<<<(BB * NN * (NN / 4)) / 256, 256, 0, stream>>>(tmpl, unif, m);
    gs_f2bf<<<(BB * NN * FF) / 256, 256, 0, stream>>>(feature, fbf);

    // 10 Sinkhorn iterations:
    //   row softmax (applies previous iteration's inverted col stats),
    //   then rebuild col stats and invert them.
    for (int it = 0; it < ITERS; ++it) {
        gs_row_softmax<<<BB * NN, 256, 0, stream>>>(m, m, csum, it > 0 ? 1 : 0);
        gs_zero<<<(CS_N + 255) / 256, 256, 0, stream>>>(csum, CS_N);
        gs_col_stats<<<dim3(NN / 256, NN / 256, BB), 256, 0, stream>>>(m, csum);
        gs_rcp<<<(CS_N + 255) / 256, 256, 0, stream>>>(csum, CS_N);
    }

    // final col softmax fused into WMMA GEMM A-load
    gs_gemm_wmma<<<dim3(NN / 16, BB), 256, 0, stream>>>(m, csum, fbf, out);
}
